// MultiHeadAttention_58531814310232
// MI455X (gfx1250) — matrix operand order repaired and checked by reading, not compiled, not run on hardware
//
#include <hip/hip_runtime.h>
#include <hip/hip_bf16.h>

// ---------------- types ----------------
typedef __attribute__((ext_vector_type(16))) _Float16 v16h;
typedef __attribute__((ext_vector_type(8)))  float    v8f;
typedef __attribute__((ext_vector_type(4)))  _Float16 v4h;
typedef __attribute__((ext_vector_type(2)))  _Float16 v2h;

#define BB   4
#define SS   2048
#define DD   1024
#define HH   16
#define DH   64
#define NTOK (BB*SS)          // 8192
#define N3   (3*DD)           // 3072

__device__ __forceinline__ v8f wmma16(v16h a, v16h b, v8f c) {
  return __builtin_amdgcn_wmma_f32_16x16x32_f16(false, a, false, b, (short)0, c, false, false);
}

__device__ __forceinline__ float rmax16(float v) {
  v = fmaxf(v, __shfl_xor(v, 1, 32));
  v = fmaxf(v, __shfl_xor(v, 2, 32));
  v = fmaxf(v, __shfl_xor(v, 4, 32));
  v = fmaxf(v, __shfl_xor(v, 8, 32));
  return v;
}
__device__ __forceinline__ float rsum16(float v) {
  v += __shfl_xor(v, 1, 32);
  v += __shfl_xor(v, 2, 32);
  v += __shfl_xor(v, 4, 32);
  v += __shfl_xor(v, 8, 32);
  return v;
}

// A-fragment index helpers (16-bit A 16x32, ISA 7.12.2):
// element (row r in 0..15, k kk in 0..31): lane = ((kk>>3)&1)*16 + r,
// half-index i = (kk>>4)*8 + (kk&7). Fragment = 32 lanes x 16 f16 = 512 f16.

// ---------------- pack x -> A-fragments (f32 -> f16) ----------------
__global__ __launch_bounds__(256) void pack_x_kernel(const float* __restrict__ x,
                                                     _Float16* __restrict__ dst) {
  int tid = blockIdx.x * 256 + threadIdx.x;           // NTOK*256 threads
  int t  = tid >> 8;
  int k0 = (tid & 255) * 4;
  float4 v = *(const float4*)(x + (size_t)t * DD + k0);
  int tile = t >> 4, r = t & 15;
  int kblk = k0 >> 5, kk = k0 & 31;
  int ga = (kk >> 3) & 1, ii = (kk >> 4) * 8 + (kk & 7);
  _Float16* p = dst + (((size_t)(tile * 32 + kblk) * 32 + ga * 16 + r) * 16 + ii);
  v4h h; h[0] = (_Float16)v.x; h[1] = (_Float16)v.y; h[2] = (_Float16)v.z; h[3] = (_Float16)v.w;
  *(v4h*)p = h;
}

// ---------------- pack weights -> B-fragments ----------------
// B 32x16 (f16): lane L holds col n' = L&15, K = kblk*32 + (i>>3)*16 + (L>>4)*8 + (i&7).
__global__ __launch_bounds__(256) void pack_w_kernel(const float* __restrict__ w,
                                                     _Float16* __restrict__ dst, int N) {
  int n = blockIdx.x * 256 + threadIdx.x;
  int k = blockIdx.y * 2;                              // K = 1024
  float a = w[(size_t)k * N + n];
  float b = w[(size_t)(k + 1) * N + n];
  int ntile = n >> 4, kblk = k >> 5;
  int ga = (k >> 3) & 1, ii = ((k >> 4) & 1) * 8 + (k & 7); // ii even
  _Float16* p = dst + (((size_t)(ntile * 32 + kblk) * 32 + ga * 16 + (n & 15)) * 16 + ii);
  v2h h; h[0] = (_Float16)a; h[1] = (_Float16)b;
  *(v2h*)p = h;
}

// ---------------- wave-tile GEMM core: 32(M) x 64(N), K=1024 ----------------
// Software-pipelined: fragments for step k+1 are issued before the 8-WMMA
// burst for step k, so each wait covers only the previous iteration's loads
// while six new b128 pairs stay in flight behind the WMMAs.
__device__ __forceinline__ void gemm_core(const _Float16* __restrict__ Af,
                                          const _Float16* __restrict__ Bf,
                                          int Moff, int Noff, int lane,
                                          v8f (&acc)[2][4]) {
  const _Float16* pa = Af + (size_t)(Moff >> 4) * 32 * 512 + lane * 16;
  const _Float16* pb = Bf + (size_t)(Noff >> 4) * 32 * 512 + lane * 16;

  v16h a0 = *(const v16h*)(pa);
  v16h a1 = *(const v16h*)(pa + (size_t)32 * 512);
  v16h b0 = *(const v16h*)(pb);
  v16h b1 = *(const v16h*)(pb + (size_t)32 * 512);
  v16h b2 = *(const v16h*)(pb + (size_t)64 * 512);
  v16h b3 = *(const v16h*)(pb + (size_t)96 * 512);

#pragma unroll 2
  for (int kb = 0; kb < 31; ++kb) {
    int nk = kb + 1;
    v16h na0 = *(const v16h*)(pa + (size_t)nk * 512);
    v16h na1 = *(const v16h*)(pa + (size_t)(32 + nk) * 512);
    v16h nb0 = *(const v16h*)(pb + (size_t)(0 * 32 + nk) * 512);
    v16h nb1 = *(const v16h*)(pb + (size_t)(1 * 32 + nk) * 512);
    v16h nb2 = *(const v16h*)(pb + (size_t)(2 * 32 + nk) * 512);
    v16h nb3 = *(const v16h*)(pb + (size_t)(3 * 32 + nk) * 512);

    acc[0][0] = wmma16(a0, b0, acc[0][0]);
    acc[1][0] = wmma16(a1, b0, acc[1][0]);
    acc[0][1] = wmma16(a0, b1, acc[0][1]);
    acc[1][1] = wmma16(a1, b1, acc[1][1]);
    acc[0][2] = wmma16(a0, b2, acc[0][2]);
    acc[1][2] = wmma16(a1, b2, acc[1][2]);
    acc[0][3] = wmma16(a0, b3, acc[0][3]);
    acc[1][3] = wmma16(a1, b3, acc[1][3]);

    a0 = na0; a1 = na1;
    b0 = nb0; b1 = nb1; b2 = nb2; b3 = nb3;
  }
  // final k-step
  acc[0][0] = wmma16(a0, b0, acc[0][0]);
  acc[1][0] = wmma16(a1, b0, acc[1][0]);
  acc[0][1] = wmma16(a0, b1, acc[0][1]);
  acc[1][1] = wmma16(a1, b1, acc[1][1]);
  acc[0][2] = wmma16(a0, b2, acc[0][2]);
  acc[1][2] = wmma16(a1, b2, acc[1][2]);
  acc[0][3] = wmma16(a0, b3, acc[0][3]);
  acc[1][3] = wmma16(a1, b3, acc[1][3]);
}

// ---------------- QKV projection ----------------
__global__ __launch_bounds__(256) void gemm_qkv_kernel(
    const _Float16* __restrict__ Af, const _Float16* __restrict__ Bf,
    const float* __restrict__ bias,
    float* __restrict__ kout, float* __restrict__ vout,
    _Float16* __restrict__ qfrag, _Float16* __restrict__ kf16,
    _Float16* __restrict__ vt16) {
  int tid = threadIdx.x, wid = tid >> 5, lane = tid & 31;
  int g = lane >> 4, c = lane & 15;
  int Moff = blockIdx.x * 64 + (wid >> 2) * 32;
  int Noff = blockIdx.y * 256 + (wid & 3) * 64;
  v8f acc[2][4] = {};
  gemm_core(Af, Bf, Moff, Noff, lane, acc);

#pragma unroll
  for (int nt = 0; nt < 4; ++nt) {
    int n = Noff + nt * 16 + c;
    float bv = bias[n];
    int region = n >> 10, nn = n & 1023;
    int h = nn >> 6, d = nn & 63;
#pragma unroll
    for (int mt = 0; mt < 2; ++mt) {
#pragma unroll
      for (int j = 0; j < 8; ++j) {
        int t = Moff + mt * 16 + g * 8 + j;
        float val = acc[mt][nt][j] + bv;
        int b_ = t >> 11, s = t & (SS - 1);
        int bh = b_ * HH + h;
        if (region == 0) {           // Q -> A-fragment layout per (b,h)
          int kk = d & 31, dblk = d >> 5;
          int ga = kk >> 4, ii = kk & 15;
          size_t idx = (((size_t)(bh * 128 + (s >> 4)) * 2 + dblk) * 32 + ga * 16 + (s & 15)) * 16 + ii;
          qfrag[idx] = (_Float16)val;
        } else if (region == 1) {    // K -> d_out f32 + row-major f16
          size_t idx = ((size_t)(bh * SS + s)) * DH + d;
          kout[idx] = val;
          kf16[idx] = (_Float16)val;
        } else {                     // V -> d_out f32 + transposed f16 [dh][S]
          size_t idx = ((size_t)(bh * SS + s)) * DH + d;
          vout[idx] = val;
          vt16[((size_t)(bh * DH + d)) * SS + s] = (_Float16)val;
        }
      }
    }
  }
}

// ---------------- causal flash attention, one wave per 16-query tile ----------------
__global__ __launch_bounds__(256) void attn_kernel(
    const _Float16* __restrict__ qfrag, const _Float16* __restrict__ kf16,
    const _Float16* __restrict__ vt16, _Float16* __restrict__ ofrag) {
  __shared__ __align__(32) _Float16 lds[8 * 512];
  int tid = threadIdx.x, wid = tid >> 5, lane = tid & 31;
  int g = lane >> 4, c = lane & 15;
  int bh = blockIdx.x;
  int qt = blockIdx.y * 8 + wid;
  int q0 = qt * 16;

  const _Float16* qb = qfrag + ((size_t)(bh * 128 + qt) * 2) * 512 + lane * 16;
  v16h qa0 = *(const v16h*)qb;
  v16h qa1 = *(const v16h*)(qb + 512);
  const _Float16* Kb = kf16 + (size_t)bh * SS * DH;
  const _Float16* Vb = vt16 + (size_t)bh * DH * SS;
  _Float16* pl = &lds[wid * 512];

  float m[8], l[8];
#pragma unroll
  for (int j = 0; j < 8; ++j) { m[j] = -1e30f; l[j] = 0.f; }
  v8f o[4] = {};

  int nblk = (qt >> 1) + 1;
  for (int blk = 0; blk < nblk; ++blk) {
    int kb0 = blk * 32;
    v8f s0 = {}, s1 = {};
    {
      v16h b0 = *(const v16h*)(Kb + (size_t)(kb0 + c) * DH + g * 16);
      v16h b1 = *(const v16h*)(Kb + (size_t)(kb0 + 16 + c) * DH + g * 16);
      v16h b2 = *(const v16h*)(Kb + (size_t)(kb0 + c) * DH + 32 + g * 16);
      v16h b3 = *(const v16h*)(Kb + (size_t)(kb0 + 16 + c) * DH + 32 + g * 16);
      s0 = wmma16(qa0, b0, s0);
      s1 = wmma16(qa0, b1, s1);
      s0 = wmma16(qa1, b2, s0);
      s1 = wmma16(qa1, b3, s1);
    }
    bool mask0 = (kb0 + 15) > q0;
    bool mask1 = (kb0 + 31) > q0;
#pragma unroll
    for (int j = 0; j < 8; ++j) {
      int qrow = q0 + g * 8 + j;
      float x0 = s0[j] * 0.125f;
      float x1 = s1[j] * 0.125f;
      x0 = (mask0 && (kb0 + c) > qrow) ? -1e30f : x0;
      x1 = (mask1 && (kb0 + 16 + c) > qrow) ? -1e30f : x1;
      float bm = rmax16(fmaxf(x0, x1));
      float mn = fmaxf(m[j], bm);
      float alpha = __expf(m[j] - mn);
      float p0 = __expf(x0 - mn);
      float p1 = __expf(x1 - mn);
      l[j] = l[j] * alpha + rsum16(p0 + p1);
      m[j] = mn;
      o[0][j] *= alpha; o[1][j] *= alpha; o[2][j] *= alpha; o[3][j] *= alpha;
      int r = g * 8 + j;
      int la = r;                             // contiguous-k A lane: kk=c in lane r, kk=16+c in lane 16+r
      pl[la * 16 + c]           = (_Float16)p0;
      pl[(16 + la) * 16 + c]    = (_Float16)p1;
    }
    asm volatile("" ::: "memory");            // DS in-order within wave
    v16h pa = *(const v16h*)(pl + lane * 16);
#pragma unroll
    for (int nt = 0; nt < 4; ++nt) {
      v16h vb = *(const v16h*)(Vb + (size_t)(nt * 16 + c) * SS + kb0 + g * 16);
      o[nt] = wmma16(pa, vb, o[nt]);
    }
    asm volatile("" ::: "memory");
  }

  int b_ = bh >> 4, h = bh & 15;
#pragma unroll
  for (int j = 0; j < 8; ++j) {
    int r = g * 8 + j;
    int t = b_ * SS + q0 + r;
    float inv = 1.0f / l[j];
#pragma unroll
    for (int nt = 0; nt < 4; ++nt) {
      int n = h * DH + nt * 16 + c;
      float val = o[nt][j] * inv;
      int kk = n & 31, kblk = n >> 5;
      int ga = (kk >> 3) & 1, ii = (kk >> 4) * 8 + (kk & 7);
      ofrag[(((size_t)(t >> 4) * 32 + kblk) * 32 + ga * 16 + r) * 16 + ii] = (_Float16)val;
    }
  }
}

// ---------------- output projection ----------------
__global__ __launch_bounds__(256) void gemm_out_kernel(
    const _Float16* __restrict__ Af, const _Float16* __restrict__ Bf,
    const float* __restrict__ bias, float* __restrict__ out) {
  int tid = threadIdx.x, wid = tid >> 5, lane = tid & 31;
  int g = lane >> 4, c = lane & 15;
  int Moff = blockIdx.x * 64 + (wid >> 2) * 32;
  int Noff = blockIdx.y * 256 + (wid & 3) * 64;
  v8f acc[2][4] = {};
  gemm_core(Af, Bf, Moff, Noff, lane, acc);
#pragma unroll
  for (int nt = 0; nt < 4; ++nt) {
    int n = Noff + nt * 16 + c;
    float bv = bias[n];
#pragma unroll
    for (int mt = 0; mt < 2; ++mt) {
#pragma unroll
      for (int j = 0; j < 8; ++j) {
        int t = Moff + mt * 16 + g * 8 + j;
        out[(size_t)t * DD + n] = acc[mt][nt][j] + bv;
      }
    }
  }
}

// ---------------- launcher ----------------
extern "C" void kernel_launch(void* const* d_in, const int* in_sizes, int n_in,
                              void* d_out, int out_size, void* d_ws, size_t ws_size,
                              hipStream_t stream) {
  const float* x     = (const float*)d_in[0];
  const float* w_in  = (const float*)d_in[1];
  const float* b_in  = (const float*)d_in[2];
  const float* w_out = (const float*)d_in[3];
  const float* b_out = (const float*)d_in[4];

  float* outp = (float*)d_out;                         // [B,S,D]
  float* kout = outp + (size_t)NTOK * DD;              // [B,H,S,dh]
  float* vout = kout + (size_t)NTOK * DD;              // [B,H,S,dh]

  char* ws = (char*)d_ws;
  size_t off = 0;
  _Float16* Xfrag   = (_Float16*)(ws + off); off += (size_t)NTOK * DD * 2;       // 16 MB
  _Float16* WinFrag = (_Float16*)(ws + off); off += (size_t)DD * N3 * 2;         // 6 MB
  _Float16* WoutFrag= (_Float16*)(ws + off); off += (size_t)DD * DD * 2;         // 2 MB
  _Float16* Qfrag   = (_Float16*)(ws + off); off += (size_t)NTOK * DD * 2;       // 16 MB
  _Float16* Kf16    = (_Float16*)(ws + off); off += (size_t)NTOK * DD * 2;       // 16 MB
  _Float16* Vt16    = (_Float16*)(ws + off); off += (size_t)NTOK * DD * 2;       // 16 MB
  _Float16* Ofrag   = (_Float16*)(ws + off); off += (size_t)NTOK * DD * 2;       // 16 MB

  pack_x_kernel<<<NTOK, 256, 0, stream>>>(x, Xfrag);
  pack_w_kernel<<<dim3(N3 / 256, DD / 2), 256, 0, stream>>>(w_in, WinFrag, N3);
  pack_w_kernel<<<dim3(DD / 256, DD / 2), 256, 0, stream>>>(w_out, WoutFrag, DD);
  gemm_qkv_kernel<<<dim3(NTOK / 64, N3 / 256), 256, 0, stream>>>(
      Xfrag, WinFrag, b_in, kout, vout, Qfrag, Kf16, Vt16);
  attn_kernel<<<dim3(BB * HH, SS / 16 / 8), 256, 0, stream>>>(Qfrag, Kf16, Vt16, Ofrag);
  gemm_out_kernel<<<dim3(NTOK / 64, DD / 256), 256, 0, stream>>>(
      Ofrag, WoutFrag, b_out, outp);
}
